// MultiHeadAttention_87282325389866
// MI455X (gfx1250) — matrix operand order repaired and checked by reading, not compiled, not run on hardware
//
#include <hip/hip_runtime.h>
#include <hip/hip_bf16.h>

// ---------------------------------------------------------------------------
// Types for CDNA5 WMMA (wave32): v_wmma_f32_16x16x32_bf16
// ---------------------------------------------------------------------------
typedef __attribute__((ext_vector_type(16))) __bf16 v16bf;
typedef __attribute__((ext_vector_type(8)))  __bf16 v8bf;
typedef __attribute__((ext_vector_type(4)))  __bf16 v4bf;
typedef __attribute__((ext_vector_type(8)))  float  v8f;

union BF16x16 { v16bf v; v8bf h[2]; };

static __device__ __forceinline__ v8f wmma_bf16(v16bf a, v16bf b, v8f c) {
    // 8 args: (neg_a, A, neg_b, B, c_mod, C, reuse_a, reuse_b)
    return __builtin_amdgcn_wmma_f32_16x16x32_bf16(false, a, false, b,
                                                   (short)0, c, false, false);
}

// A-matrix fragment, 16x32 bf16 (ISA 7.12.2 "16-bit A-Matrix 16x32"):
//   lane r=lane&15 -> row; lane half selects K offset 0 / 8;
//   elems 0..7 -> K = koff+0..7 ; elems 8..15 -> K = 16+koff+0..7
static __device__ __forceinline__ v16bf load_fragA(const __bf16* __restrict__ base,
                                                   int ld, int row16, int k0, int lane) {
    int r    = lane & 15;
    int koff = (lane >> 4) << 3;                 // 0 or 8
    const __bf16* p = base + (size_t)(row16 + r) * ld + (k0 + koff);
    BF16x16 f;
    f.h[0] = *(const v8bf*)(p);                  // K = k0+koff .. +7
    f.h[1] = *(const v8bf*)(p + 16);             // K = k0+koff+16 .. +23
    return f.v;
}

// B-matrix fragment, 32x16 bf16 (column N per lane, lane half selects K offset 0 / 8;
// elems 0..7 -> K = koff+0..7, elems 8..15 -> K = 16+koff+0..7). Source is [N, K] row-major (NT GEMM).
static __device__ __forceinline__ v16bf load_fragB(const __bf16* __restrict__ base,
                                                   int ld, int n16, int k0, int lane) {
    int n  = lane & 15;
    int ks = k0 + ((lane >> 4) << 3);            // 0 or 8
    const __bf16* p = base + (size_t)(n16 + n) * ld + ks;
    BF16x16 f;
    f.h[0] = *(const v8bf*)(p);                  // K = ks .. ks+7
    f.h[1] = *(const v8bf*)(p + 16);             // K = ks+16 .. ks+23
    return f.v;
}

// ---------------------------------------------------------------------------
// Problem constants
// ---------------------------------------------------------------------------
#define HDIM   1024
#define NHEAD  16
#define DHEAD  64
#define BATCH  2
#define SEQ    2048
#define MTOT   (BATCH * SEQ)   // 4096 tokens

// ---------------------------------------------------------------------------
// fp32 -> bf16 conversion (4-wide)
// ---------------------------------------------------------------------------
__global__ void cvt_f32_bf16_x4(const float* __restrict__ s,
                                __bf16* __restrict__ d, int n4) {
    int i = blockIdx.x * blockDim.x + threadIdx.x;
    if (i < n4) {
        float4 f = ((const float4*)s)[i];
        v4bf o = { (__bf16)f.x, (__bf16)f.y, (__bf16)f.z, (__bf16)f.w };
        ((v4bf*)d)[i] = o;
    }
}

// ---------------------------------------------------------------------------
// QKV projection: C[t, o] = sum_k x[t,k] * W[o,k] + b[o]   (NT GEMM, K=1024)
// z = 0/1/2 selects Q/K/V. Q,K stored [b,h,s,d]; V stored transposed [b,h,d,s].
// Register-blocked: each wave computes a 64(M) x 32(N) macro-tile
//   = 4x2 WMMA tiles: per K-step 6 fragment loads feed 8 WMMAs.
// ---------------------------------------------------------------------------
__global__ void qkv_gemm(const __bf16* __restrict__ X,
                         const __bf16* __restrict__ Wq,
                         const __bf16* __restrict__ Wk,
                         const __bf16* __restrict__ Wv,
                         const float* __restrict__ bq,
                         const float* __restrict__ bk,
                         const float* __restrict__ bv,
                         __bf16* __restrict__ Qb,
                         __bf16* __restrict__ Kb,
                         __bf16* __restrict__ Vt) {
    const int lane  = threadIdx.x & 31;
    const int warp  = threadIdx.x >> 5;
    const int which = blockIdx.z;
    const int m0  = blockIdx.x * 64;                  // rows   [m0,  m0+64)
    const int n0  = (blockIdx.y * 4 + warp) * 32;     // cols   [n0,  n0+32)
    const __bf16* W    = (which == 0) ? Wq : (which == 1) ? Wk : Wv;
    const float*  bias = (which == 0) ? bq : (which == 1) ? bk : bv;

    v8f c[4][2] = {};
    for (int k0 = 0; k0 < HDIM; k0 += 32) {
        v16bf a[4], b[2];
#pragma unroll
        for (int i = 0; i < 4; ++i)
            a[i] = load_fragA(X, HDIM, m0 + i * 16, k0, lane);
#pragma unroll
        for (int j = 0; j < 2; ++j)
            b[j] = load_fragB(W, HDIM, n0 + j * 16, k0, lane);
#pragma unroll
        for (int i = 0; i < 4; ++i)
#pragma unroll
            for (int j = 0; j < 2; ++j)
                c[i][j] = wmma_bf16(a[i], b[j], c[i][j]);
    }

    const int nsub  = lane & 15;
    const int mbase = (lane >> 4) << 3;
#pragma unroll
    for (int j = 0; j < 2; ++j) {
        const int n  = n0 + j * 16 + nsub;            // output channel
        const int hh = n >> 6, dv = n & 63;
        const float bsv = bias[n];
#pragma unroll
        for (int i = 0; i < 4; ++i) {
#pragma unroll
            for (int r = 0; r < 8; ++r) {
                int t  = m0 + i * 16 + mbase + r;      // global token
                int bb = t >> 11, ss = t & (SEQ - 1);
                __bf16 o = (__bf16)(c[i][j][r] + bsv);
                if (which == 0)
                    Qb[(((size_t)bb * NHEAD + hh) * SEQ + ss) * DHEAD + dv] = o;
                else if (which == 1)
                    Kb[(((size_t)bb * NHEAD + hh) * SEQ + ss) * DHEAD + dv] = o;
                else
                    Vt[(((size_t)bb * NHEAD + hh) * DHEAD + dv) * SEQ + ss] = o;
            }
        }
    }
}

// ---------------------------------------------------------------------------
// Flash-style attention. One wave owns a 16-query tile of one (b,h);
// streams 32 keys / iteration: 4 score WMMAs + softmax + 5 PV WMMAs.
// The probability row-sum is computed on the MATRIX pipe via an extra
// all-ones B tile (acc4 = P @ 1), so the online-softmax "l" update rides on
// the same alpha rescale as the output accumulators; only the row-max still
// uses the shuffle network. The 4 waves of a block process consecutive
// q-tiles of the SAME (b,h) so K/V fragments hit WGP$; next key tile is
// prefetched (global_prefetch_b8) while the softmax VALU work runs.
// ---------------------------------------------------------------------------
__global__ void attn_kernel(const __bf16* __restrict__ Q,
                            const __bf16* __restrict__ Kmat,
                            const __bf16* __restrict__ Vt,
                            const float* __restrict__ mask,   // [B, S]
                            __bf16* __restrict__ O) {         // [B, S, H]
    __shared__ __bf16 pshare[4][16][40];         // per-wave 16x32 P tile (+pad)
    const int lane = threadIdx.x & 31;
    const int warp = threadIdx.x >> 5;
    const int tile = blockIdx.x * 4 + warp;      // 0..4095 (warps share (b,h))
    const int qt = tile & 127;                   // query tile 0..127
    const int h  = (tile >> 7) & (NHEAD - 1);
    const int b  = tile >> 11;

    const __bf16* Qh = Q    + ((size_t)(b * NHEAD + h)) * SEQ * DHEAD;
    const __bf16* Kh = Kmat + ((size_t)(b * NHEAD + h)) * SEQ * DHEAD;
    const __bf16* Vh = Vt   + ((size_t)(b * NHEAD + h)) * DHEAD * SEQ;
    const float*  mk = mask + (size_t)b * SEQ;

    // Q fragments: 16 rows x 64 head-dim -> two K=32 A-fragments (kept resident)
    const v16bf qa0 = load_fragA(Qh, DHEAD, qt * 16, 0,  lane);
    const v16bf qa1 = load_fragA(Qh, DHEAD, qt * 16, 32, lane);

    // constant all-ones B tile (B[k][n] = 1 regardless of layout)
    v16bf ones;
#pragma unroll
    for (int i = 0; i < 16; ++i) ones[i] = (__bf16)1.0f;

    float mrun[8];
#pragma unroll
    for (int r = 0; r < 8; ++r) mrun[r] = -1e30f;
    v8f acc0 = {}, acc1 = {}, acc2 = {}, acc3 = {};
    v8f acc4 = {};                               // running row-sum l (P @ ones)

    const float scale = 0.125f;                  // 1/sqrt(64)
    const int nsub = lane & 15;
    const int mb   = (lane >> 4) << 3;

    for (int j = 0; j < SEQ; j += 32) {
        // ---- scores for 32 keys (two 16-key C tiles) ----
        v16bf kb0a = load_fragB(Kh, DHEAD, j,      0,  lane);
        v16bf kb0b = load_fragB(Kh, DHEAD, j,      32, lane);
        v16bf kb1a = load_fragB(Kh, DHEAD, j + 16, 0,  lane);
        v16bf kb1b = load_fragB(Kh, DHEAD, j + 16, 32, lane);
        v8f s0 = {}, s1 = {};
        s0 = wmma_bf16(qa0, kb0a, s0);
        s0 = wmma_bf16(qa1, kb0b, s0);
        s1 = wmma_bf16(qa0, kb1a, s1);
        s1 = wmma_bf16(qa1, kb1b, s1);

        // prefetch next key tile while softmax VALU runs (speculative, WGP scope)
        if (j + 32 < SEQ) {
            __builtin_prefetch(Kh + (size_t)(j + 32 + lane) * DHEAD, 0, 3);
            __builtin_prefetch(Vh + (size_t)(2 * lane) * SEQ + (j + 32), 0, 3);
            __builtin_prefetch(Vh + (size_t)(2 * lane + 1) * SEQ + (j + 32), 0, 3);
        }

        const float mv0 = mk[j + nsub];
        const float mv1 = mk[j + 16 + nsub];
        float sm0[8], sm1[8], alpha[8];
#pragma unroll
        for (int r = 0; r < 8; ++r) {
            sm0[r] = s0[r] * scale + mv0;
            sm1[r] = s1[r] * scale + mv1;
        }
        // ---- online softmax: row-max via shuffle net, exp per element ----
#pragma unroll
        for (int r = 0; r < 8; ++r) {
            float rm = fmaxf(sm0[r], sm1[r]);
            rm = fmaxf(rm, __shfl_xor(rm, 1, 32));
            rm = fmaxf(rm, __shfl_xor(rm, 2, 32));
            rm = fmaxf(rm, __shfl_xor(rm, 4, 32));
            rm = fmaxf(rm, __shfl_xor(rm, 8, 32));
            float mnew = fmaxf(mrun[r], rm);
            alpha[r] = __expf(mrun[r] - mnew);
            mrun[r]  = mnew;
            sm0[r] = __expf(sm0[r] - mnew);
            sm1[r] = __expf(sm1[r] - mnew);
        }
#pragma unroll
        for (int r = 0; r < 8; ++r) {
            acc0[r] *= alpha[r]; acc1[r] *= alpha[r];
            acc2[r] *= alpha[r]; acc3[r] *= alpha[r];
            acc4[r] *= alpha[r];                 // l = l * alpha ...
        }
        // ---- C-layout P -> A-fragment layout via LDS ----
#pragma unroll
        for (int r = 0; r < 8; ++r) {
            pshare[warp][mb + r][nsub]      = (__bf16)sm0[r];
            pshare[warp][mb + r][16 + nsub] = (__bf16)sm1[r];
        }
        __syncthreads();
        v16bf pa;
        {
            int rr = lane & 15;
            int ko = (lane >> 4) << 3;
            BF16x16 f;
            f.h[0] = *(const v8bf*)(&pshare[warp][rr][ko]);
            f.h[1] = *(const v8bf*)(&pshare[warp][rr][ko + 16]);
            pa = f.v;
        }
        __syncthreads();
        // ---- PV: 4 dv tiles + ones tile (row-sum), K=32 keys ----
        v16bf vb0 = load_fragB(Vh, SEQ, 0,  j, lane);
        v16bf vb1 = load_fragB(Vh, SEQ, 16, j, lane);
        v16bf vb2 = load_fragB(Vh, SEQ, 32, j, lane);
        v16bf vb3 = load_fragB(Vh, SEQ, 48, j, lane);
        acc0 = wmma_bf16(pa, vb0, acc0);
        acc1 = wmma_bf16(pa, vb1, acc1);
        acc2 = wmma_bf16(pa, vb2, acc2);
        acc3 = wmma_bf16(pa, vb3, acc3);
        acc4 = wmma_bf16(pa, ones, acc4);        // ... + rowsum(P)
    }

    // ---- normalize and store (token-major, heads concatenated) ----
#pragma unroll
    for (int r = 0; r < 8; ++r) {
        float inv = 1.0f / acc4[r];              // every lane of the row holds l
        int ss = qt * 16 + mb + r;
        size_t row = ((size_t)b * SEQ + ss) * HDIM + h * DHEAD;
        O[row + nsub]      = (__bf16)(acc0[r] * inv);
        O[row + 16 + nsub] = (__bf16)(acc1[r] * inv);
        O[row + 32 + nsub] = (__bf16)(acc2[r] * inv);
        O[row + 48 + nsub] = (__bf16)(acc3[r] * inv);
    }
}

// ---------------------------------------------------------------------------
// Output projection: out[t, o] = sum_k A[t,k] * Wo[o,k] + bo[o]  (f32 out)
// Same 64x32 register-blocked structure as qkv_gemm.
// ---------------------------------------------------------------------------
__global__ void oproj_gemm(const __bf16* __restrict__ A,
                           const __bf16* __restrict__ W,
                           const float* __restrict__ bias,
                           float* __restrict__ Cout) {
    const int lane = threadIdx.x & 31;
    const int warp = threadIdx.x >> 5;
    const int m0 = blockIdx.x * 64;
    const int n0 = (blockIdx.y * 4 + warp) * 32;

    v8f c[4][2] = {};
    for (int k0 = 0; k0 < HDIM; k0 += 32) {
        v16bf a[4], b[2];
#pragma unroll
        for (int i = 0; i < 4; ++i)
            a[i] = load_fragA(A, HDIM, m0 + i * 16, k0, lane);
#pragma unroll
        for (int j = 0; j < 2; ++j)
            b[j] = load_fragB(W, HDIM, n0 + j * 16, k0, lane);
#pragma unroll
        for (int i = 0; i < 4; ++i)
#pragma unroll
            for (int j = 0; j < 2; ++j)
                c[i][j] = wmma_bf16(a[i], b[j], c[i][j]);
    }

    const int nsub  = lane & 15;
    const int mbase = (lane >> 4) << 3;
#pragma unroll
    for (int j = 0; j < 2; ++j) {
        const int n = n0 + j * 16 + nsub;
        const float bsv = bias[n];
#pragma unroll
        for (int i = 0; i < 4; ++i)
#pragma unroll
            for (int r = 0; r < 8; ++r)
                Cout[(size_t)(m0 + i * 16 + mbase + r) * HDIM + n] = c[i][j][r] + bsv;
    }
}

// ---------------------------------------------------------------------------
// Launch
// ---------------------------------------------------------------------------
extern "C" void kernel_launch(void* const* d_in, const int* in_sizes, int n_in,
                              void* d_out, int out_size, void* d_ws, size_t ws_size,
                              hipStream_t stream) {
    const float* x    = (const float*)d_in[0];
    const float* mask = (const float*)d_in[1];
    const float* Wq   = (const float*)d_in[2];
    const float* bq   = (const float*)d_in[3];
    const float* Wk   = (const float*)d_in[4];
    const float* bk   = (const float*)d_in[5];
    const float* Wv   = (const float*)d_in[6];
    const float* bv   = (const float*)d_in[7];
    const float* Wo   = (const float*)d_in[8];
    const float* bo   = (const float*)d_in[9];
    float* out = (float*)d_out;

    // Workspace partition (bf16 elements): ~50 MB total
    __bf16* ws  = (__bf16*)d_ws;
    const size_t MH = (size_t)MTOT * HDIM;       // 4096*1024
    const size_t HH = (size_t)HDIM * HDIM;
    __bf16* xbf = ws;
    __bf16* wqb = xbf + MH;
    __bf16* wkb = wqb + HH;
    __bf16* wvb = wkb + HH;
    __bf16* wob = wvb + HH;
    __bf16* Qb  = wob + HH;
    __bf16* Kb  = Qb + MH;
    __bf16* Vtb = Kb + MH;
    __bf16* Ab  = Vtb + MH;

    // 1) fp32 -> bf16
    {
        int n4 = (int)(MH / 4);
        cvt_f32_bf16_x4<<<(n4 + 255) / 256, 256, 0, stream>>>(x, xbf, n4);
        int w4 = (int)(HH / 4);
        int g  = (w4 + 255) / 256;
        cvt_f32_bf16_x4<<<g, 256, 0, stream>>>(Wq, wqb, w4);
        cvt_f32_bf16_x4<<<g, 256, 0, stream>>>(Wk, wkb, w4);
        cvt_f32_bf16_x4<<<g, 256, 0, stream>>>(Wv, wvb, w4);
        cvt_f32_bf16_x4<<<g, 256, 0, stream>>>(Wo, wob, w4);
    }
    // 2) QKV projections: M=4096 -> 64 macro-rows, N=1024 -> 8 blocks of 4x32
    qkv_gemm<<<dim3(MTOT / 64, HDIM / 128, 3), 128, 0, stream>>>(
        xbf, wqb, wkb, wvb, bq, bk, bv, Qb, Kb, Vtb);
    // 3) attention: 4096 (b,h,qtile) wave-tiles, 4 per block sharing (b,h)
    attn_kernel<<<dim3((BATCH * NHEAD * (SEQ / 16)) / 4), 128, 0, stream>>>(
        Qb, Kb, Vtb, mask, Ab);
    // 4) output projection (f32 result)
    oproj_gemm<<<dim3(MTOT / 64, HDIM / 128), 128, 0, stream>>>(
        Ab, wob, bo, out);
}
